// BipolarCellNetwork_82841329205927
// MI455X (gfx1250) — matrix operand order repaired and checked by reading, not compiled, not run on hardware
//
#include <hip/hip_runtime.h>
#include <hip/hip_bf16.h>

typedef __attribute__((ext_vector_type(16))) _Float16 v16h;
typedef __attribute__((ext_vector_type(8)))  float    v8f;

#define NBC 14
#define NAC 45
#define KF  52
#define KW  52          // AC kernel taps (taps 52..63 zero-padded)
#define BATCH 64
#define TT  4096
#define RHO 0.05f
#define KKDIM 896       // NBC*64, 28 WMMA k-iters of 32
#define NKI   28
#define NACPAD 48       // 3 WMMA N-tiles
#define RING 128        // duplicated 64-deep ring

// output offsets (floats): y_bcn, fb, ac_out, y_lnr  (each (B,T,.))
#define O_YBCN 0
#define O_FB   (BATCH*TT*NBC)                    // 3,670,016
#define O_AC   (2*BATCH*TT*NBC)                  // 7,340,032
#define O_YLNR (2*BATCH*TT*NBC + BATCH*TT*NAC)   // 19,136,512

// ---------------- LDS layout (bytes) for bcn kernel -----------------
#define SM_WLS   0                       // 48*896 f16      = 86016
#define SM_HIST  86016                   // 16*14*128 f16   = 57344
#define SM_DRIVE 143360                  // 16*48 f32       = 3072
#define SM_AC    146432                  // 16*48 f32       = 3072
#define SM_POOL  149504                  // 16*14 f32       = 896
#define SM_WBC   150400                  // 14*45 f32       = 2520
#define SM_PARM  152920                  // slope14,off14,osc14,acsl45,acof45
#define SM_TOTAL 153600

// ---------------------------------------------------------------
// 1) Fused AC-kernel x BC->AC weight: Wglob[a][n*64+kl] (f16), kl>=52 -> 0
//    ack[a,k] = exp(-kt/td) - exp(-((td+tr)/(td*tr))*kt), L2-normalized
// ---------------------------------------------------------------
__global__ void bcn_prep_kernel(const float* __restrict__ lbaw,   // (45,14)
                                const float* __restrict__ ltr,    // (45)
                                const float* __restrict__ ltd,    // (45)
                                _Float16* __restrict__ Wglob) {   // (48,896)
    int a = threadIdx.x;
    if (a >= NACPAD) return;
    if (a < NAC) {
        float tr = __expf(ltr[a]);
        float td = __expf(ltd[a]);
        float c2 = (td + tr) / (td * tr);
        float kv[KW];
        float ss = 0.f;
        #pragma unroll
        for (int k = 0; k < KW; ++k) {
            float kt = 0.8f - k * (1.0f / 64.0f);
            float v  = __expf(-kt / td) - __expf(-c2 * kt);
            kv[k] = v; ss += v * v;
        }
        float inv = rsqrtf(ss);
        for (int n = 0; n < NBC; ++n) {
            float w = __expf(lbaw[a * NBC + n]) * inv;
            for (int kl = 0; kl < 64; ++kl)
                Wglob[a * KKDIM + n * 64 + kl] =
                    (kl < KW) ? (_Float16)(w * kv[kl]) : (_Float16)0.f;
        }
    } else {
        for (int kk = 0; kk < KKDIM; ++kk) Wglob[a * KKDIM + kk] = (_Float16)0.f;
    }
}

// ---------------------------------------------------------------
// 2) Causal FIR: xf[t*896 + b*14 + n] = sum_j x[b, t-51+j]*bc_kernels[n,j]
// ---------------------------------------------------------------
__global__ void bcn_conv_kernel(const float* __restrict__ x,
                                const float* __restrict__ bk,
                                float* __restrict__ xf) {
    int id = blockIdx.x * blockDim.x + threadIdx.x;
    if (id >= TT * BATCH * NBC) return;
    int n  = id % NBC;
    int bt = id / NBC;
    int b  = bt % BATCH;
    int t  = bt / BATCH;
    const float* xb = x + b * TT;
    const float* w  = bk + n * KF;
    float acc = 0.f;
    #pragma unroll
    for (int j = 0; j < KF; ++j) {
        int ti = t - (KF - 1) + j;
        if (ti >= 0) acc += xb[ti] * w[j];
    }
    xf[id] = acc;
}

// ---------------------------------------------------------------
// 3) LNR branch: independent scan per (b,n) — 896 threads
// ---------------------------------------------------------------
__global__ void bcn_lnr_kernel(const float* __restrict__ xf,
                               const float* __restrict__ lss,
                               const float* __restrict__ soff,
                               const float* __restrict__ osc,
                               float* __restrict__ out_lnr) {
    int id = blockIdx.x * blockDim.x + threadIdx.x;
    if (id >= BATCH * NBC) return;
    int n = id % NBC, b = id / NBC;
    float slope = __expf(lss[n]);
    float off   = soff[n];
    float sc    = osc[n];
    float pool  = 1.f;
    for (int t = 0; t < TT; ++t) {
        float v     = xf[t * (BATCH * NBC) + id];
        float p     = 1.f / (1.f + __expf(-slope * (v - off)));
        float pool2 = pool + (1.f - pool) * RHO;
        float rel   = p * pool2;
        pool        = pool2 - rel;
        out_lnr[((size_t)b * TT + t) * NBC + n] = sc * rel;
    }
}

// ---------------------------------------------------------------
// 4) BCN recurrence. 4 blocks x 128 threads; block = 16 batches.
//    Per step: drive = [16 x 896]f16 @ [896 x 48]f16 via 28x
//    v_wmma_f32_16x16x32_f16 per N-tile (waves 0..2), fully unrolled so
//    every A element is an unconditional ds_load_u16 with immediate offset.
// ---------------------------------------------------------------
__global__ __launch_bounds__(128)
void bcn_step_kernel(const float* __restrict__ xf,
                     const _Float16* __restrict__ Wglob,
                     const float* __restrict__ lss,    // (14)
                     const float* __restrict__ soff,   // (14)
                     const float* __restrict__ lacb,   // (14,45) log_ac_bc_weight
                     const float* __restrict__ lacs,   // (45)
                     const float* __restrict__ acof,   // (45)
                     const float* __restrict__ osc,    // (14)
                     float* __restrict__ out) {
    extern __shared__ char smem[];
    _Float16* Wls   = (_Float16*)(smem + SM_WLS);    // [48][896]  B operand
    _Float16* hist  = (_Float16*)(smem + SM_HIST);   // [16][14][128] dup ring
    float* driveLS  = (float*)(smem + SM_DRIVE);     // [16][48]
    float* acLS     = (float*)(smem + SM_AC);        // [16][48]
    float* poolLS   = (float*)(smem + SM_POOL);      // [16][14]
    float* wbcLS    = (float*)(smem + SM_WBC);       // [14][45]  (-exp)
    float* slopeLS  = (float*)(smem + SM_PARM);
    float* offLS    = slopeLS + NBC;
    float* oscLS    = offLS   + NBC;
    float* acslLS   = oscLS   + NBC;                 // 45
    float* acofLS   = acslLS  + NAC;                 // 45

    const int tid = threadIdx.x;
    const int wgb = blockIdx.x;          // batch tile: global batch = wgb*16 + lb

    // ---- init: weights -> LDS, zero ring, params ----
    for (int i = tid; i < (NACPAD * KKDIM * 2) / 4; i += 128)
        ((int*)(smem + SM_WLS))[i] = ((const int*)Wglob)[i];
    for (int i = tid; i < (16 * NBC * RING * 2) / 4; i += 128)
        ((int*)(smem + SM_HIST))[i] = 0;
    for (int i = tid; i < 16 * NBC; i += 128) poolLS[i] = 1.0f;
    for (int i = tid; i < NBC * NAC; i += 128) wbcLS[i] = -__expf(lacb[i]);
    if (tid < NBC) { slopeLS[tid] = __expf(lss[tid]); offLS[tid] = soff[tid]; oscLS[tid] = osc[tid]; }
    if (tid < NAC) { acslLS[tid] = __expf(lacs[tid]); acofLS[tid] = acof[tid]; }
    __syncthreads();

    const int lane = tid & 31;
    const int wv   = tid >> 5;
    const int m    = lane & 15;     // A row (batch) / B col (AC)
    const int hi   = lane >> 4;

    // per-lane invariant bases
    const _Float16* wrow  = Wls + (wv * 16 + m) * KKDIM + hi * 8;
    const _Float16* hbase = hist + m * (NBC * RING) + hi * 8;

    for (int t = 0; t < TT; ++t) {
        // ---------- phase A: WMMA drive GEMM (waves 0..2, EXEC all-1) ----------
        if (wv < 3) {
            const int tq = (t + 12) & 63;          // ring phase, scalar per step
            const _Float16* hrow = hbase + tq;     // all A offsets now constants
            v8f c = {0.f, 0.f, 0.f, 0.f, 0.f, 0.f, 0.f, 0.f};
            #pragma unroll
            for (int ki = 0; ki < NKI; ++ki) {
                v16h a;
                #pragma unroll
                for (int j = 0; j < 16; ++j) {
                    // kk = ki*32 + hi*8 + cj ; n = ki>>1 ; kl = (ki&1)*32 + hi*8 + cj
                    const int cj  = (j < 8) ? j : (j + 8);
                    const int off = (ki >> 1) * RING + (ki & 1) * 32 + cj;
                    a[j] = hrow[off];
                }
                typedef __attribute__((ext_vector_type(8))) _Float16 v8h_;
                v16h b;
                b.lo = *(const v8h_*)(wrow + ki * 32);
                b.hi = *(const v8h_*)(wrow + ki * 32 + 16);
                c = __builtin_amdgcn_wmma_f32_16x16x32_f16(
                        false, a, false, b, (short)0, c, false, false);
            }
            #pragma unroll
            for (int r = 0; r < 8; ++r)
                driveLS[(r + 8 * hi) * NACPAD + wv * 16 + m] = c[r];
        }
        __syncthreads();

        // ---------- phase B: AC sigmoid ----------
        for (int i = tid; i < 16 * NAC; i += 128) {
            int lb = i / NAC, a = i - lb * NAC;
            float d  = driveLS[lb * NACPAD + a];
            float ao = 1.f / (1.f + __expf(-acslLS[a] * (d - acofLS[a])));
            acLS[lb * NACPAD + a] = ao;
            int gb = wgb * 16 + lb;
            out[O_AC + ((size_t)gb * TT + t) * NAC + a] = ao;
        }
        __syncthreads();

        // ---------- phase C: feedback, release, pool update, ring write ----------
        for (int i = tid; i < 16 * NBC; i += 128) {
            int lb = i / NBC, n = i - lb * NBC;
            float fb = 0.f;
            #pragma unroll
            for (int a2 = 0; a2 < NAC; ++a2)
                fb += acLS[lb * NACPAD + a2] * wbcLS[n * NAC + a2];
            int gb = wgb * 16 + lb;
            float xv    = xf[(size_t)t * (BATCH * NBC) + gb * NBC + n];
            float prob  = 1.f / (1.f + __expf(-slopeLS[n] * (xv + fb - offLS[n])));
            float pool  = poolLS[lb * NBC + n];
            float pool2 = pool + (1.f - pool) * RHO;
            float rel   = prob * pool2;
            poolLS[lb * NBC + n] = pool2 - rel;
            _Float16 h = (_Float16)rel;
            int s0 = t & 63;                       // tap entering the window
            hist[(lb * NBC + n) * RING + s0]      = h;
            hist[(lb * NBC + n) * RING + s0 + 64] = h;   // duplicate (no wrap on read)
            out[O_YBCN + ((size_t)gb * TT + t) * NBC + n] = oscLS[n] * rel;
            out[O_FB   + ((size_t)gb * TT + t) * NBC + n] = fb;
        }
        __syncthreads();
    }
}

// ---------------------------------------------------------------
extern "C" void kernel_launch(void* const* d_in, const int* in_sizes, int n_in,
                              void* d_out, int out_size, void* d_ws, size_t ws_size,
                              hipStream_t stream) {
    const float* x    = (const float*)d_in[0];   // (64,4096)
    const float* bck  = (const float*)d_in[1];   // (14,52)
    const float* lss  = (const float*)d_in[2];   // (14)
    const float* soff = (const float*)d_in[3];   // (14)
    const float* lbaw = (const float*)d_in[4];   // (45,14)
    const float* ltr  = (const float*)d_in[5];   // (45)
    const float* ltd  = (const float*)d_in[6];   // (45)
    const float* lacs = (const float*)d_in[7];   // (45)
    const float* acof = (const float*)d_in[8];   // (45)
    const float* lacb = (const float*)d_in[9];   // (14,45)
    const float* osc  = (const float*)d_in[10];  // (14)
    float* out = (float*)d_out;

    // workspace: [Wglob f16 48*896 = 86016 B][xf f32 T*B*NBC]
    _Float16* Wglob = (_Float16*)d_ws;
    float* xf = (float*)((char*)d_ws + NACPAD * KKDIM * 2);

    bcn_prep_kernel<<<1, 64, 0, stream>>>(lbaw, ltr, ltd, Wglob);

    int convN = TT * BATCH * NBC;
    bcn_conv_kernel<<<(convN + 255) / 256, 256, 0, stream>>>(x, bck, xf);

    bcn_lnr_kernel<<<4, 256, 0, stream>>>(xf, lss, soff, osc, out + O_YLNR);

    bcn_step_kernel<<<4, 128, SM_TOTAL, stream>>>(xf, Wglob, lss, soff, lacb,
                                                  lacs, acof, osc, out);
}
